// AttentionLayer_43224550868107
// MI455X (gfx1250) — compile-verified, matrix operand order checked by reading, not run on hardware
//
#include <hip/hip_runtime.h>
#include <hip/hip_bf16.h>
#include <math.h>

typedef _Float16 v16h __attribute__((ext_vector_type(16)));
typedef _Float16 v8h  __attribute__((ext_vector_type(8)));
typedef float    v8f  __attribute__((ext_vector_type(8)));
typedef float    v4f  __attribute__((ext_vector_type(4)));
typedef int      v4i  __attribute__((ext_vector_type(4)));

#define NN     8192
#define DIN    512
#define DOUT   256
#define ALPHA  0.2f

__device__ __forceinline__ float lrelu(float x) { return x > 0.f ? x : ALPHA * x; }

// ---------------------------------------------------------------------------
// Kernel P: weight [512,256] f32 row-major -> wT [256,512] f16 (transposed)
// ---------------------------------------------------------------------------
__global__ void gat_prep_wT(const float* __restrict__ weight, _Float16* __restrict__ wT) {
    int idx = blockIdx.x * 256 + threadIdx.x;       // 131072 total
    int n = idx & 255;
    int k = idx >> 8;
    wT[(size_t)n * DIN + k] = (_Float16)weight[(size_t)k * DOUT + n];
}

// ---------------------------------------------------------------------------
// Kernel 1: v = node @ W   (WMMA f16, f32 accumulate)
// block = 256 thr (8 waves); block computes 64 rows x 256 cols.
// wave w: M-tile = w&3 (16 rows), N-half = w>>2 (8 N-tiles of 16).
// Writes v (f32 row-major) and vT (f16, [DOUT, NN]).
// ---------------------------------------------------------------------------
__global__ __launch_bounds__(256)
void gat_gemm1(const float* __restrict__ node, const _Float16* __restrict__ wT,
               float* __restrict__ v, _Float16* __restrict__ vT) {
    int tid  = threadIdx.x;
    int w    = tid >> 5, lane = tid & 31;
    int mt   = w & 3,    nh   = w >> 2;
    int half  = lane >> 4;
    int half8 = half << 3;
    int l15   = lane & 15;

    int tileRow = blockIdx.x * 64 + mt * 16;
    int m       = tileRow + l15;
    int nbase   = nh * 128 + l15;

    v8f acc[8];
#pragma unroll
    for (int t = 0; t < 8; ++t) acc[t] = (v8f)(0.f);

    const float* arow = node + (size_t)m * DIN;

    for (int kb = 0; kb < DIN; kb += 32) {
        v4f a0 = *(const v4f*)(arow + kb + half8);
        v4f a1 = *(const v4f*)(arow + kb + half8 + 4);
        v4f a2 = *(const v4f*)(arow + kb + 16 + half8);
        v4f a3 = *(const v4f*)(arow + kb + 16 + half8 + 4);
        v16h Af;
#pragma unroll
        for (int i = 0; i < 4; ++i) {
            Af[i]      = (_Float16)a0[i];
            Af[4 + i]  = (_Float16)a1[i];
            Af[8 + i]  = (_Float16)a2[i];
            Af[12 + i] = (_Float16)a3[i];
        }
#pragma unroll
        for (int t = 0; t < 8; ++t) {
            const _Float16* bp = wT + (size_t)(nbase + t * 16) * DIN + kb;
            v8h b0 = *(const v8h*)(bp + half8);
            v8h b1 = *(const v8h*)(bp + 16 + half8);
            v16h Bf;
#pragma unroll
            for (int i = 0; i < 8; ++i) { Bf[i] = b0[i]; Bf[8 + i] = b1[i]; }
            acc[t] = __builtin_amdgcn_wmma_f32_16x16x32_f16(
                false, Af, false, Bf, (short)0, acc[t], false, false);
        }
    }

#pragma unroll
    for (int t = 0; t < 8; ++t) {
        int n = nbase + t * 16;
        v8h hv;
#pragma unroll
        for (int r = 0; r < 8; ++r) {
            float val = acc[t][r];
            v[(size_t)(tileRow + half8 + r) * DOUT + n] = val;
            hv[r] = (_Float16)val;
        }
        *(v8h*)(vT + (size_t)n * NN + tileRow + half8) = hv;
    }
}

// ---------------------------------------------------------------------------
// Kernel QK: Q[i] = v[i,:].a[0:256]; K = v[i,:].a[256:512];
// also KD[i] = (e^K, e^{0.2K}) table used by both attention passes.
// one wave per row
// ---------------------------------------------------------------------------
__global__ __launch_bounds__(256)
void gat_qk(const float* __restrict__ v, const float* __restrict__ avec,
            float* __restrict__ Qarr, float2* __restrict__ KD) {
    int w = threadIdx.x >> 5, lane = threadIdx.x & 31;
    int row = blockIdx.x * 8 + w;
    int c = lane * 8;
    const float* vr = v + (size_t)row * DOUT;
    v4f x0 = *(const v4f*)(vr + c);
    v4f x1 = *(const v4f*)(vr + c + 4);
    v4f q0 = *(const v4f*)(avec + c);
    v4f q1 = *(const v4f*)(avec + c + 4);
    v4f k0 = *(const v4f*)(avec + DOUT + c);
    v4f k1 = *(const v4f*)(avec + DOUT + c + 4);
    float q = 0.f, k = 0.f;
#pragma unroll
    for (int i = 0; i < 4; ++i) {
        q += x0[i] * q0[i] + x1[i] * q1[i];
        k += x0[i] * k0[i] + x1[i] * k1[i];
    }
#pragma unroll
    for (int off = 16; off; off >>= 1) {
        q += __shfl_xor(q, off, 32);
        k += __shfl_xor(k, off, 32);
    }
    if (lane == 0) {
        Qarr[row] = q;
        KD[row]   = make_float2(__expf(k), __expf(ALPHA * k));
    }
}

// ---------------------------------------------------------------------------
// Kernel A: single streaming pass over adj (non-temporal b128 loads).
// Packs 1-bit mask (interleaved convention: for each 128-entry group g,
// word g*4+e holds entry g*128 + 4*l + e at bit l) and accumulates the
// unshifted softmax denominator S0 = sum_masked e^{lrelu(Q_i+K_j)} using
// the factorized KD table. One wave per row.
// ---------------------------------------------------------------------------
__global__ __launch_bounds__(256)
void gat_mask_s0(const int* __restrict__ adj, const float* __restrict__ Qarr,
                 const float2* __restrict__ KDg, unsigned* __restrict__ maskw,
                 float* __restrict__ invS) {
    __shared__ float2 KDs[NN];      // 64 KB
    int tid = threadIdx.x;
    for (int i = tid; i < NN; i += 256) KDs[i] = KDg[i];
    __syncthreads();

    int w = tid >> 5, lane = tid & 31;
    int row = blockIdx.x * 8 + w;
    float Qi = Qarr[row];
    float cp = __expf(Qi);            // e^Q   (positive branch)
    float cn = __expf(ALPHA * Qi);    // e^{0.2Q}
    float tq = __expf(-Qi);           // threshold: d1 > tq  <=>  Q+K > 0
    const int* arow = adj + (size_t)row * NN;

    float s = 0.f;
    for (int it = 0; it < 64; ++it) {
        int base = it * 128 + lane * 4;
        v4i av = __builtin_nontemporal_load((const v4i*)(arow + base));
        unsigned b0 = (unsigned)__ballot(av[0] > 0);
        unsigned b1 = (unsigned)__ballot(av[1] > 0);
        unsigned b2 = (unsigned)__ballot(av[2] > 0);
        unsigned b3 = (unsigned)__ballot(av[3] > 0);
        if (lane == 0)
            *(uint4*)(maskw + (size_t)row * 256 + it * 4) = make_uint4(b0, b1, b2, b3);

        const v4f* dp = (const v4f*)(KDs + base);   // 4 float2 = 2 x v4f
        v4f p0 = dp[0], p1 = dp[1];
#pragma unroll
        for (int e = 0; e < 4; ++e) {
            float d1 = (e < 2) ? p0[(e & 1) * 2]     : p1[(e & 1) * 2];
            float d2 = (e < 2) ? p0[(e & 1) * 2 + 1] : p1[(e & 1) * 2 + 1];
            bool  cg = d1 > tq;
            float val = (cg ? cp : cn) * (cg ? d1 : d2);
            s += (av[e] > 0) ? val : 0.f;
        }
    }
#pragma unroll
    for (int off = 16; off; off >>= 1) s += __shfl_xor(s, off, 32);
    if (lane == 0) invS[row] = (s > 0.f) ? 1.f / s : 0.f;
}

// ---------------------------------------------------------------------------
// Kernel B: out = softmaxP @ v via WMMA. Block = 64 rows x 256 cols, 8 waves:
// wave w owns M-tile (w&3, 16 rows) x all 256 cols (16 acc tiles) over the
// j-half (w>>2)*4096 .. +4096  -> no duplicated P-reconstruction work.
// j-half-1 waves dump accumulators to LDS; j-half-0 waves combine, then run
// the fused leaky_relu + row-L2-normalize + bias epilogue fully in-wave.
// ---------------------------------------------------------------------------
__global__ __launch_bounds__(256)
void gat_attn(const _Float16* __restrict__ vT, const float* __restrict__ Qarr,
              const float* __restrict__ invS, const float2* __restrict__ KDg,
              const unsigned* __restrict__ maskw, const float* __restrict__ bias,
              float* __restrict__ out) {
    __shared__ float2 KDs[NN];                  // 64 KB
    __shared__ float  accLDS[4][16][32][8];     // 64 KB
    int tid = threadIdx.x;
    for (int i = tid; i < NN; i += 256) KDs[i] = KDg[i];
    __syncthreads();

    int w = tid >> 5, lane = tid & 31;
    int mt = w & 3, jh = w >> 2;
    int half  = lane >> 4;
    int half8 = half << 3;
    int l15   = lane & 15;

    int tileRow = blockIdx.x * 64 + mt * 16;
    int m       = tileRow + l15;

    float Qm = Qarr[m];
    float iS = invS[m];
    float c1 = __expf(Qm) * iS;          // positive-branch row constant
    float c2 = __expf(ALPHA * Qm) * iS;  // negative-branch row constant
    float tq = __expf(-Qm);              // branch threshold on d1
    const unsigned* mrow = maskw + (size_t)m * 256;

    v8f acc[16];
#pragma unroll
    for (int t = 0; t < 16; ++t) acc[t] = (v8f)(0.f);

    int jbeg = jh * (NN / 2);
    for (int jb128 = jbeg; jb128 < jbeg + NN / 2; jb128 += 128) {
        uint4 mw = *(const uint4*)(mrow + (jb128 >> 7) * 4);
#pragma unroll
        for (int cch = 0; cch < 4; ++cch) {
            int jb = jb128 + cch * 32;
            unsigned sh = (unsigned)(8 * cch + 2 * half);
            unsigned wsh[4] = { mw.x >> sh, mw.y >> sh, mw.z >> sh, mw.w >> sh };

            const v4f* pA = (const v4f*)(KDs + jb + half8);        // run 0
            const v4f* pB = (const v4f*)(KDs + jb + 16 + half8);   // run 1
            v4f ra[4], rb[4];
#pragma unroll
            for (int u = 0; u < 4; ++u) { ra[u] = pA[u]; rb[u] = pB[u]; }

            v16h Af;
#pragma unroll
            for (int i = 0; i < 8; ++i) {
                float d1a = ra[i >> 1][(i & 1) * 2];
                float d2a = ra[i >> 1][(i & 1) * 2 + 1];
                float d1b = rb[i >> 1][(i & 1) * 2];
                float d2b = rb[i >> 1][(i & 1) * 2 + 1];
                bool  cA = d1a > tq;
                bool  cB = d1b > tq;
                unsigned we = wsh[i & 3];
                bool  bA = (we >> (i >> 2)) & 1u;
                bool  bB = (we >> (4 + (i >> 2))) & 1u;
                float fA = bA ? (cA ? c1 : c2) : 0.f;
                float fB = bB ? (cB ? c1 : c2) : 0.f;
                Af[i]     = (_Float16)(fA * (cA ? d1a : d2a));
                Af[8 + i] = (_Float16)(fB * (cB ? d1b : d2b));
            }
#pragma unroll
            for (int t = 0; t < 16; ++t) {
                const _Float16* vp = vT + (size_t)(t * 16 + l15) * NN + jb;
                v8h b0 = *(const v8h*)(vp + half8);
                v8h b1 = *(const v8h*)(vp + 16 + half8);
                v16h Bf;
#pragma unroll
                for (int i = 0; i < 8; ++i) { Bf[i] = b0[i]; Bf[8 + i] = b1[i]; }
                acc[t] = __builtin_amdgcn_wmma_f32_16x16x32_f16(
                    false, Af, false, Bf, (short)0, acc[t], false, false);
            }
        }
    }

    // combine j-halves: upper-half waves export, lower-half waves reduce
    if (jh == 1) {
#pragma unroll
        for (int t = 0; t < 16; ++t) {
            v4f lo = { acc[t][0], acc[t][1], acc[t][2], acc[t][3] };
            v4f hi = { acc[t][4], acc[t][5], acc[t][6], acc[t][7] };
            *(v4f*)(&accLDS[mt][t][lane][0]) = lo;
            *(v4f*)(&accLDS[mt][t][lane][4]) = hi;
        }
    }
    __syncthreads();
    if (jh == 0) {
#pragma unroll
        for (int t = 0; t < 16; ++t) {
            v4f lo = *(const v4f*)(&accLDS[mt][t][lane][0]);
            v4f hi = *(const v4f*)(&accLDS[mt][t][lane][4]);
#pragma unroll
            for (int k = 0; k < 4; ++k) {
                acc[t][k]     += lo[k];
                acc[t][4 + k] += hi[k];
            }
        }

        // leaky_relu + per-row sum of squares (rows fully in-wave now)
        float sq[8];
#pragma unroll
        for (int r = 0; r < 8; ++r) sq[r] = 0.f;
#pragma unroll
        for (int t = 0; t < 16; ++t)
#pragma unroll
            for (int r = 0; r < 8; ++r) {
                float val = lrelu(acc[t][r]);
                acc[t][r] = val;
                sq[r] += val * val;
            }
#pragma unroll
        for (int off = 8; off; off >>= 1)
#pragma unroll
            for (int r = 0; r < 8; ++r) sq[r] += __shfl_xor(sq[r], off, 32);

        float bn[16];
#pragma unroll
        for (int t = 0; t < 16; ++t) bn[t] = bias[t * 16 + l15];

#pragma unroll
        for (int r = 0; r < 8; ++r) {
            float scale = 1.f / fmaxf(sqrtf(sq[r]), 1e-12f);
            size_t ob   = (size_t)(tileRow + half8 + r) * DOUT;
#pragma unroll
            for (int t = 0; t < 16; ++t)
                out[ob + t * 16 + l15] = acc[t][r] * scale + bn[t];
        }
    }
}

// ---------------------------------------------------------------------------
extern "C" void kernel_launch(void* const* d_in, const int* in_sizes, int n_in,
                              void* d_out, int out_size, void* d_ws, size_t ws_size,
                              hipStream_t stream) {
    const float* node   = (const float*)d_in[0];
    const int*   adj    = (const int*)d_in[1];
    const float* weight = (const float*)d_in[2];
    const float* avec   = (const float*)d_in[3];
    const float* bias   = (const float*)d_in[4];
    float* out = (float*)d_out;

    char* ws = (char*)d_ws;
    float*    v    = (float*)(ws);                               // 8 MB
    _Float16* vT   = (_Float16*)(ws + 8388608);                  // 4 MB
    _Float16* wT   = (_Float16*)(ws + 12582912);                 // 256 KB
    float*    Qarr = (float*)(ws + 12845056);                    // 32 KB
    float2*   KD   = (float2*)(ws + 12910592);                   // 64 KB
    float*    invS = (float*)(ws + 12976128);                    // 32 KB
    unsigned* mask = (unsigned*)(ws + 13008896);                 // 8 MB

    gat_prep_wT <<<512,  256, 0, stream>>>(weight, wT);
    gat_gemm1   <<<128,  256, 0, stream>>>(node, wT, v, vT);
    gat_qk      <<<1024, 256, 0, stream>>>(v, avec, Qarr, KD);
    gat_mask_s0 <<<1024, 256, 0, stream>>>(adj, Qarr, KD, mask, invS);
    gat_attn    <<<128,  256, 0, stream>>>(vT, Qarr, invS, KD, mask, bias, out);
}
